// OurGAL_37185826849364
// MI455X (gfx1250) — matrix operand order repaired and checked by reading, not compiled, not run on hardware
//
#include <hip/hip_runtime.h>
#include <math.h>

// ---------------------------------------------------------------------------
// Types for CDNA5 WMMA (gfx1250, wave32)
// ---------------------------------------------------------------------------
typedef __attribute__((ext_vector_type(16))) __bf16         v16bf;
typedef __attribute__((ext_vector_type(8)))  float          v8f;

#define DIM 128
#define AGE_K 7
#define OCC_K 21
#define EPSV 1e-5f
#define STATS_BLOCKS 240

// ---------------------------------------------------------------------------
// 0) zero scratch (agg + degi + bce region)
// ---------------------------------------------------------------------------
__global__ void k_zero(unsigned int* __restrict__ p, long long n) {
  long long i  = (long long)blockIdx.x * blockDim.x + threadIdx.x;
  long long st = (long long)gridDim.x * blockDim.x;
  for (; i < n; i += st) p[i] = 0u;
}

// ---------------------------------------------------------------------------
// 1) column stats over E [nent x 128]: per-block partial sum / sumsq
// ---------------------------------------------------------------------------
__global__ void k_colstats_partial(const float* __restrict__ E, int nent,
                                   float* __restrict__ psum, float* __restrict__ psq) {
  int c = threadIdx.x;  // 0..127, coalesced across one row
  float s = 0.f, q = 0.f;
  for (int r = blockIdx.x; r < nent; r += gridDim.x) {
    float v = E[(long long)r * DIM + c];
    s += v;
    q += v * v;
  }
  psum[blockIdx.x * DIM + c] = s;
  psq [blockIdx.x * DIM + c] = q;
}

// 2) finalize: mu, var -> scale = gamma*rsqrt(var+eps), shift = beta - mu*scale
__global__ void k_colstats_final(const float* __restrict__ psum, const float* __restrict__ psq,
                                 int nblocks, float inv_n,
                                 const float* __restrict__ gamma, const float* __restrict__ beta,
                                 float* __restrict__ scale, float* __restrict__ shift) {
  int c = threadIdx.x;
  float s = 0.f, q = 0.f;
  for (int b = 0; b < nblocks; ++b) { s += psum[b * DIM + c]; q += psq[b * DIM + c]; }
  float mu   = s * inv_n;
  float var  = q * inv_n - mu * mu;
  float rstd = rsqrtf(var + EPSV);
  float sc   = gamma[c] * rstd;
  scale[c] = sc;
  shift[c] = beta[c] - mu * sc;
}

// ---------------------------------------------------------------------------
// 3) edge scatter: agg[dst] += normalize(E[src]); degi[dst] += 1
//    one wave32 per edge; lane handles 4 contiguous columns (float4)
// ---------------------------------------------------------------------------
__global__ void __launch_bounds__(256) k_edge_scatter(
    const float* __restrict__ E, const int* __restrict__ src, const int* __restrict__ dst,
    const float* __restrict__ scale, const float* __restrict__ shift,
    float* __restrict__ agg, int* __restrict__ degi, int ne) {
  int lane = threadIdx.x & 31;
  int e = blockIdx.x * 8 + (threadIdx.x >> 5);
  if (e >= ne) return;
  int s = src[e];
  int d = dst[e];
  const float4 ev = ((const float4*)(E + (long long)s * DIM))[lane];
  const float4 sc = ((const float4*)scale)[lane];
  const float4 sh = ((const float4*)shift)[lane];
  float4 v;
  v.x = ev.x * sc.x + sh.x;
  v.y = ev.y * sc.y + sh.y;
  v.z = ev.z * sc.z + sh.z;
  v.w = ev.w * sc.w + sh.w;
  float* ap = agg + (long long)d * DIM + lane * 4;
  __hip_atomic_fetch_add(ap + 0, v.x, __ATOMIC_RELAXED, __HIP_MEMORY_SCOPE_AGENT);
  __hip_atomic_fetch_add(ap + 1, v.y, __ATOMIC_RELAXED, __HIP_MEMORY_SCOPE_AGENT);
  __hip_atomic_fetch_add(ap + 2, v.z, __ATOMIC_RELAXED, __HIP_MEMORY_SCOPE_AGENT);
  __hip_atomic_fetch_add(ap + 3, v.w, __ATOMIC_RELAXED, __HIP_MEMORY_SCOPE_AGENT);
  if (lane == 0)
    __hip_atomic_fetch_add(degi + d, 1, __ATOMIC_RELAXED, __HIP_MEMORY_SCOPE_AGENT);
}

// ---------------------------------------------------------------------------
// 4) gathered GEMM via v_wmma_f32_16x16x32_bf16:
//    u[b,:] = relu((agg[neighbor[b],:]/deg) @ W_gnn + b_gnn), only 8192 rows.
//    Block = 8 waves; wave w owns N-tile [16w,16w+16); block owns one 16-row
//    M-tile. W_gnn transposed to LDS as bf16 [n][k].
// ---------------------------------------------------------------------------
__global__ void __launch_bounds__(256) k_gemm_wmma(
    const float* __restrict__ agg, const int* __restrict__ degi,
    const int* __restrict__ neighbor,
    const float* __restrict__ Wg,    // [k][n] row-major, 128x128
    const float* __restrict__ bgnn,  // [128]
    float* __restrict__ u, int batch) {
  __shared__ __bf16 Wt[DIM * DIM];  // bf16, [n][k]
  for (int i = threadIdx.x; i < DIM * DIM; i += blockDim.x) {
    int k = i >> 7, n = i & (DIM - 1);      // coalesced read of W row-major
    Wt[n * DIM + k] = (__bf16)Wg[i];        // hw convert (v_cvt_pk_bf16_f32)
  }
  __syncthreads();

  int lane = threadIdx.x & 31;
  int wave = threadIdx.x >> 5;
  int hl   = (lane >> 4) & 1;       // 0: lanes 0-15, 1: lanes 16-31
  int n0   = wave * 16;
  int mt   = blockIdx.x;

  int m = mt * 16 + (lane & 15);
  if (m >= batch) m = batch - 1;    // clamp (keeps EXEC all-ones for WMMA)
  int node = neighbor[m];
  float invd = 1.0f / fmaxf((float)degi[node], 1.0f);
  const float* rowp = agg + (long long)node * DIM;

  int n = n0 + (lane & 15);
  float bias = bgnn[n];
  v8f c;
#pragma unroll
  for (int r = 0; r < 8; ++r) c[r] = bias;

#pragma unroll
  for (int kc = 0; kc < 4; ++kc) {
    int kbase = kc * 32;
    // ---- A fragment (16x32 bf16): lane row = lane&15
    // lanes<16: K kbase+0..15; lanes>=16: K kbase+16..31
    int abase = kbase + hl * 16;
    float4 a0 = *(const float4*)(rowp + abase);
    float4 a1 = *(const float4*)(rowp + abase + 4);
    float4 a2 = *(const float4*)(rowp + abase + 8);
    float4 a3 = *(const float4*)(rowp + abase + 12);
    v16bf A;
    A[0]  = (__bf16)(a0.x * invd); A[1]  = (__bf16)(a0.y * invd);
    A[2]  = (__bf16)(a0.z * invd); A[3]  = (__bf16)(a0.w * invd);
    A[4]  = (__bf16)(a1.x * invd); A[5]  = (__bf16)(a1.y * invd);
    A[6]  = (__bf16)(a1.z * invd); A[7]  = (__bf16)(a1.w * invd);
    A[8]  = (__bf16)(a2.x * invd); A[9]  = (__bf16)(a2.y * invd);
    A[10] = (__bf16)(a2.z * invd); A[11] = (__bf16)(a2.w * invd);
    A[12] = (__bf16)(a3.x * invd); A[13] = (__bf16)(a3.y * invd);
    A[14] = (__bf16)(a3.z * invd); A[15] = (__bf16)(a3.w * invd);
    // ---- B fragment (32x16 bf16): lane col = lane&15,
    // lanes<16: K kbase+0..15; lanes>=16: K kbase+16..31 (contiguous in Wt[n][k])
    int bbase = kbase + hl * 16;
    v16bf B = *(const v16bf*)(&Wt[n * DIM + bbase]);
    c = __builtin_amdgcn_wmma_f32_16x16x32_bf16(
        /*neg_a=*/false, A, /*neg_b=*/false, B,
        /*c_mod=*/(short)0, c, /*reuse_a=*/false, /*reuse_b=*/false);
  }

  // C/D layout: VGPR r -> row (r + 8*hl), col = lane&15
  int rbase = mt * 16 + hl * 8;
#pragma unroll
  for (int r = 0; r < 8; ++r) {
    int row = rbase + r;
    if (row < batch) {
      float v = c[r] > 0.f ? c[r] : 0.f;
      u[(long long)row * DIM + n] = v;
    }
  }
}

// ---------------------------------------------------------------------------
// 5) heads: one wave32 per sample; shfl_xor cross-lane dot reductions
// ---------------------------------------------------------------------------
__global__ void __launch_bounds__(256) k_heads(
    const float* __restrict__ u, const int* __restrict__ gender,
    const float* __restrict__ Wgv, const float* __restrict__ bgp,
    const float* __restrict__ Wage, const float* __restrict__ bage,
    const float* __restrict__ Wocc, const float* __restrict__ bocc,
    float* __restrict__ out_age, float* __restrict__ out_g, float* __restrict__ out_occ,
    float* __restrict__ bce, int batch) {
  __shared__ float sWg[DIM];
  __shared__ float sWa[AGE_K * DIM];   // transposed [j][k]
  __shared__ float sWo[OCC_K * DIM];   // transposed [j][k]
  __shared__ float sba[AGE_K];
  __shared__ float sbo[OCC_K];
  for (int i = threadIdx.x; i < DIM; i += blockDim.x) sWg[i] = Wgv[i];
  for (int i = threadIdx.x; i < AGE_K * DIM; i += blockDim.x) {
    int j = i % AGE_K, k = i / AGE_K;
    sWa[j * DIM + k] = Wage[i];
  }
  for (int i = threadIdx.x; i < OCC_K * DIM; i += blockDim.x) {
    int j = i % OCC_K, k = i / OCC_K;
    sWo[j * DIM + k] = Wocc[i];
  }
  if (threadIdx.x < AGE_K) sba[threadIdx.x] = bage[threadIdx.x];
  if (threadIdx.x < OCC_K) sbo[threadIdx.x] = bocc[threadIdx.x];
  __syncthreads();

  int lane   = threadIdx.x & 31;
  int gwave  = blockIdx.x * (blockDim.x >> 5) + (threadIdx.x >> 5);
  int nwaves = gridDim.x * (blockDim.x >> 5);
  float bg = bgp[0];

  for (int b = gwave; b < batch; b += nwaves) {
    float4 u4 = ((const float4*)(u + (long long)b * DIM))[lane];
    // gender logit + BCE term
    float4 w4 = ((const float4*)sWg)[lane];
    float p = u4.x * w4.x + u4.y * w4.y + u4.z * w4.z + u4.w * w4.w;
    for (int off = 16; off; off >>= 1) p += __shfl_xor(p, off, 32);
    float gp = p + bg;
    if (lane == 0) {
      out_g[b] = gp;
      float z = (float)gender[b];
      bce[b] = fmaxf(gp, 0.f) - gp * z + log1pf(expf(-fabsf(gp)));
    }
#pragma unroll
    for (int j = 0; j < AGE_K; ++j) {
      float4 a4 = ((const float4*)(sWa + j * DIM))[lane];
      float q = u4.x * a4.x + u4.y * a4.y + u4.z * a4.z + u4.w * a4.w;
      for (int off = 16; off; off >>= 1) q += __shfl_xor(q, off, 32);
      if (lane == 0) out_age[b * AGE_K + j] = q + sba[j];
    }
#pragma unroll
    for (int j = 0; j < OCC_K; ++j) {
      float4 o4 = ((const float4*)(sWo + j * DIM))[lane];
      float q = u4.x * o4.x + u4.y * o4.y + u4.z * o4.z + u4.w * o4.w;
      for (int off = 16; off; off >>= 1) q += __shfl_xor(q, off, 32);
      if (lane == 0) out_occ[b * OCC_K + j] = q + sbo[j];
    }
  }
}

// ---------------------------------------------------------------------------
// 6) deterministic BCE mean reduction -> d_out[0]
// ---------------------------------------------------------------------------
__global__ void k_bce_reduce(const float* __restrict__ bce, int batch, float* __restrict__ out) {
  __shared__ float s[256];
  float a = 0.f;
  for (int i = threadIdx.x; i < batch; i += 256) a += bce[i];
  s[threadIdx.x] = a;
  __syncthreads();
  for (int w = 128; w; w >>= 1) {
    if ((int)threadIdx.x < w) s[threadIdx.x] += s[threadIdx.x + w];
    __syncthreads();
  }
  if (threadIdx.x == 0) out[0] = s[0] / (float)batch;
}

// ---------------------------------------------------------------------------
// launch
// ---------------------------------------------------------------------------
extern "C" void kernel_launch(void* const* d_in, const int* in_sizes, int n_in,
                              void* d_out, int out_size, void* d_ws, size_t ws_size,
                              hipStream_t stream) {
  const float* E     = (const float*)d_in[0];
  const float* gamma = (const float*)d_in[1];
  const float* beta  = (const float*)d_in[2];
  const float* W_gnn = (const float*)d_in[3];
  const float* b_gnn = (const float*)d_in[4];
  const float* W_g   = (const float*)d_in[5];
  const float* b_g   = (const float*)d_in[6];
  const float* W_age = (const float*)d_in[7];
  const float* b_age = (const float*)d_in[8];
  const float* W_occ = (const float*)d_in[9];
  const float* b_occ = (const float*)d_in[10];
  const int*   src   = (const int*)d_in[11];
  const int*   dst   = (const int*)d_in[12];
  const int*   nbr   = (const int*)d_in[13];
  const int*   gen   = (const int*)d_in[14];

  int nent  = in_sizes[0] / DIM;
  int ne    = in_sizes[11];
  int batch = in_sizes[13];

  // workspace layout (floats)
  float* ws = (float*)d_ws;
  long long off = 0;
  float* agg  = ws + off; off += (long long)nent * DIM;   // 51.2 MB
  int*   degi = (int*)(ws + off); off += nent;            // 0.4 MB
  float* bce  = ws + off; off += batch;                   // 32 KB
  long long zero_elems = off;                             // zeroed each call
  float* psum  = ws + off; off += STATS_BLOCKS * DIM;
  float* psq   = ws + off; off += STATS_BLOCKS * DIM;
  float* scale = ws + off; off += DIM;
  float* shift = ws + off; off += DIM;
  float* u     = ws + off; off += (long long)batch * DIM; // 4 MB

  k_zero<<<2048, 256, 0, stream>>>((unsigned int*)d_ws, zero_elems);
  k_colstats_partial<<<STATS_BLOCKS, DIM, 0, stream>>>(E, nent, psum, psq);
  k_colstats_final<<<1, DIM, 0, stream>>>(psum, psq, STATS_BLOCKS, 1.0f / (float)nent,
                                          gamma, beta, scale, shift);
  k_edge_scatter<<<(ne + 7) / 8, 256, 0, stream>>>(E, src, dst, scale, shift, agg, degi, ne);

  int mtiles = (batch + 15) / 16;
  k_gemm_wmma<<<mtiles, 256, 0, stream>>>(agg, degi, nbr, W_gnn, b_gnn, u, batch);

  float* out     = (float*)d_out;
  float* out_age = out + 1;
  float* out_g   = out + 1 + (long long)batch * AGE_K;
  float* out_occ = out_g + batch;
  k_heads<<<64, 256, 0, stream>>>(u, gen, W_g, b_g, W_age, b_age, W_occ, b_occ,
                                  out_age, out_g, out_occ, bce, batch);
  k_bce_reduce<<<1, 256, 0, stream>>>(bce, batch, out);
}
